// StyleGAN2Block_43971875177100
// MI455X (gfx1250) — matrix operand order repaired and checked by reading, not compiled, not run on hardware
//
#include <hip/hip_runtime.h>
#include <hip/hip_bf16.h>
#include <stdint.h>

// ---------------------------------------------------------------------------
// StyleGAN2 block for MI455X (gfx1250, wave32, WMMA).
// Memory-bound (~310 GFLOP vs tens-of-µs of HBM traffic at 23.3 TB/s) ->
//   * f16 operands, f32 accumulation: v_wmma_f32_16x16x32_f16
//   * f16 zero-padded intermediate activations (halves bandwidth, kills
//     border bounds-checks in the hot loop)
//   * weights repacked once to f16 [tap][o][ci] so the A-tile stage is a
//     pure 16B-aligned copy done with global_load_async_to_lds_b128
//     (ASYNCcnt) overlapping the synchronous B-tile transpose stage.
// ---------------------------------------------------------------------------

typedef __attribute__((ext_vector_type(8)))  float    v8f;
typedef __attribute__((ext_vector_type(8)))  _Float16 v8h;
typedef __attribute__((ext_vector_type(16))) _Float16 v16h;

#define BATCH 8
#define CH    512            // Cin == Cout == NZ == 512
#define NPIX  4096           // 64*64 output resolution
#define PROW  66             // padded row (1px zero border each side)
#define PPLANE (PROW * PROW) // 4356 elements per padded channel plane
#define LDK   40             // 32 + 8 halfword pad: 80B row stride (5*16B),
                             // conflict-free 16B-aligned ds_load_b128 frags

// ---------------------------------------------------------------------------
// zero u32 words (clears padded activation buffers incl. borders each call)
// ---------------------------------------------------------------------------
__global__ void zero_kernel(uint32_t* __restrict__ p, int n) {
    int idx = blockIdx.x * blockDim.x + threadIdx.x;
    if (idx < n) p[idx] = 0u;
}

// ---------------------------------------------------------------------------
// one-time repack: Wh[s][o][ci] = (f16) W[o][ci][s]   (s = 3*kh+kw tap)
// ---------------------------------------------------------------------------
__global__ void repack_w(const float* __restrict__ W, _Float16* __restrict__ Wh) {
    int idx = blockIdx.x * blockDim.x + threadIdx.x;     // 9 * CH * CH
    if (idx >= 9 * CH * CH) return;
    int s  = idx >> 18;                                  // CH*CH == 1<<18
    int r  = idx & (CH * CH - 1);
    int o  = r >> 9, ci = r & 511;
    Wh[idx] = (_Float16)W[((size_t)o * CH + ci) * 9 + s];
}

// ---------------------------------------------------------------------------
// wsq[o*CH+ci] = sum_k W[o,ci,k]^2   (used by demod)
// ---------------------------------------------------------------------------
__global__ void wsq_kernel(const float* __restrict__ W, float* __restrict__ wsq) {
    int idx = blockIdx.x * blockDim.x + threadIdx.x;     // CH*CH
    if (idx >= CH * CH) return;
    const float* p = W + (size_t)idx * 9;
    float a = 0.f;
#pragma unroll
    for (int k = 0; k < 9; ++k) a += p[k] * p[k];
    wsq[idx] = a;
}

// ---------------------------------------------------------------------------
// s[b*CH+c] = dot(w[b,:], sW[c,:]) + sB[c]
// ---------------------------------------------------------------------------
__global__ void style_kernel(const float* __restrict__ wlat, const float* __restrict__ sW,
                             const float* __restrict__ sB, float* __restrict__ s) {
    int idx = blockIdx.x * blockDim.x + threadIdx.x;     // BATCH*CH
    if (idx >= BATCH * CH) return;
    int b = idx >> 9, c = idx & 511;
    const float* wr = wlat + (size_t)b * CH;
    const float* sr = sW + (size_t)c * CH;
    float a = sB[c];
    for (int k = 0; k < CH; ++k) a += wr[k] * sr[k];
    s[idx] = a;
}

// ---------------------------------------------------------------------------
// d[b*CH+o] = rsqrt( sum_ci wsq[o,ci] * s[b,ci]^2 + eps )
// ---------------------------------------------------------------------------
__global__ void demod_kernel(const float* __restrict__ wsq, const float* __restrict__ s,
                             float* __restrict__ d) {
    int idx = blockIdx.x * blockDim.x + threadIdx.x;     // BATCH*CH
    if (idx >= BATCH * CH) return;
    int b = idx >> 9, o = idx & 511;
    const float* sq = wsq + (size_t)o * CH;
    const float* sr = s + (size_t)b * CH;
    float a = 1e-8f;
    for (int ci = 0; ci < CH; ++ci) { float sv = sr[ci]; a += sq[ci] * sv * sv; }
    d[idx] = rsqrtf(a);
}

// ---------------------------------------------------------------------------
// 2x bilinear upsample (align_corners=False: taps 0.25/0.75, edge clamp),
// optional per-(b,channel) style fusion. Output addressing parameterized so
// maps land in the zero-padded f16 buffer interior, rgb in a dense f32 one.
// ---------------------------------------------------------------------------
template <typename OutT>
__global__ void upsample2x_mod(const float* __restrict__ in, const float* __restrict__ style,
                               OutT* __restrict__ out, int C,
                               int oplane, int orow, int ooff) {
    int idx = blockIdx.x * blockDim.x + threadIdx.x;     // BATCH*C*NPIX
    int total = BATCH * C * NPIX;
    if (idx >= total) return;
    int p = idx & 4095, bc = idx >> 12;                  // bc = b*C + c
    int y = p >> 6, x = p & 63;
    int fy = y & 1, fx = x & 1;
    int y0 = (y >> 1) - (fy ^ 1);
    int x0 = (x >> 1) - (fx ^ 1);
    float wy0 = fy ? 0.75f : 0.25f;
    float wx0 = fx ? 0.75f : 0.25f;
    int y0c = y0 < 0 ? 0 : y0;
    int y1c = (y0 + 1 > 31) ? 31 : y0 + 1;
    int x0c = x0 < 0 ? 0 : x0;
    int x1c = (x0 + 1 > 31) ? 31 : x0 + 1;
    const float* src = in + (size_t)bc * 1024;           // 32*32 source plane
    float v = wy0 * (wx0 * src[y0c * 32 + x0c] + (1.f - wx0) * src[y0c * 32 + x1c]) +
              (1.f - wy0) * (wx0 * src[y1c * 32 + x0c] + (1.f - wx0) * src[y1c * 32 + x1c]);
    if (style) v *= style[bc];
    out[(size_t)bc * oplane + y * orow + x + ooff] = (OutT)v;
}

// ---------------------------------------------------------------------------
// Modulated 3x3 conv as implicit GEMM over 9 taps. Input pre-modulated &
// zero-padded [B,CH,66,66] f16. Block tile 128(M=out ch) x 128(N=pixels),
// 8 waves (4Mx2N), wave tile 32x64 = 2x4 WMMA accs, K-step 32.
// A tile staged with global_load_async_to_lds_b128; B tile staged with a
// synchronous u16 transpose-scatter (no bounds checks thanks to padding).
// Epilogue fuses demod, bias, noise, leaky-ReLU, optional next-layer style.
// ---------------------------------------------------------------------------
template <typename OutT>
__global__ __launch_bounds__(256) void conv3x3_wmma(
    const _Float16* __restrict__ actp,   // [B,CH,66,66] f16, padded, pre-modulated
    const _Float16* __restrict__ Whf,    // [9,CH,CH] f16 repacked weights
    const float*    __restrict__ demod,  // [B,CH]
    const float*    __restrict__ bias,   // [CH]
    const float*    __restrict__ nscale, // [CH]
    const float*    __restrict__ noise,  // [B,1,64,64]
    const float*    __restrict__ s_next, // [B,CH] or nullptr
    OutT*           __restrict__ out,    // dense or padded, see oplane/orow/ooff
    int oplane, int orow, int ooff)
{
    __shared__ alignas(16) _Float16 lds_a[128 * LDK];    // weights  [M][K]
    __shared__ alignas(16) _Float16 lds_b[128 * LDK];    // activ.   [N][K]

    const int b     = blockIdx.z;
    const int otile = blockIdx.y * 128;
    const int ptile = blockIdx.x * 128;
    const int r0    = ptile >> 6;                        // first of 2 rows
    const int tid   = threadIdx.x;
    const int wv    = tid >> 5, lane = tid & 31;
    const int wm    = wv & 3, wn = wv >> 2;              // 4 x 2 wave grid
    const int lr    = lane & 15, lh = lane >> 4;

    v8f acc[2][4];
#pragma unroll
    for (int i = 0; i < 2; ++i)
#pragma unroll
        for (int j = 0; j < 4; ++j) acc[i][j] = v8f{};

    // B-stage thread map: ci lane within chunk x 16-pixel group
    const int b_k  = tid >> 3;                           // 0..31
    const int b_pg = tid & 7;                            // 0..7
    const int b_row = r0 + (b_pg >> 2) + 1;              // padded row (pre-dy)
    const int b_col = (b_pg & 3) * 16 + 1;               // padded col (pre-dx)

    for (int shift = 0; shift < 9; ++shift) {
        const int dy = shift / 3 - 1, dx = shift % 3 - 1;
        for (int kc = 0; kc < CH; kc += 32) {
            __syncthreads();             // prev fragments fully consumed

            // ---- stage A via async DMA to LDS: 512 x 16B segments
            {
                const _Float16* abase = Whf + ((size_t)shift * CH + otile) * CH;
#pragma unroll
                for (int i = 0; i < 2; ++i) {
                    const int seg   = i * 256 + tid;     // 0..511
                    const int o_row = seg >> 2;          // 0..127
                    const int sub   = seg & 3;           // 16B unit in 64B row
                    uint32_t ldsoff = (uint32_t)(uintptr_t)&lds_a[o_row * LDK + sub * 8];
                    uint32_t goff   = (uint32_t)((o_row * CH + kc + sub * 8) * sizeof(_Float16));
                    asm volatile("global_load_async_to_lds_b128 %0, %1, %2"
                                 :
                                 : "v"(ldsoff), "v"(goff), "s"(abase)
                                 : "memory");
                }
            }
            // ---- stage B: padded act -> LDS [pixel][ci] transpose, no checks
            {
                const int ci = kc + b_k;
                const _Float16* src = actp + ((size_t)b * CH + ci) * PPLANE +
                                      (size_t)(b_row + dy) * PROW + (b_col + dx);
                _Float16* dst = &lds_b[(b_pg * 16) * LDK + b_k];
#pragma unroll
                for (int i = 0; i < 16; ++i) dst[i * LDK] = src[i];
            }
            asm volatile("s_wait_asynccnt 0x0" ::: "memory");
            __syncthreads();

            // ---- fragments per ISA 7.12.2 VGPR layouts (wave32)
            v16h afrag[2], bfrag[4];
#pragma unroll
            for (int fm = 0; fm < 2; ++fm) {
                // A 16x32: lane m=lr holds K {lh*8..+7} then {16+lh*8..+7}
                const _Float16* pa = &lds_a[(wm * 32 + fm * 16 + lr) * LDK + lh * 8];
                v8h lo = *(const v8h*)pa;
                v8h hi = *(const v8h*)(pa + 16);
                afrag[fm] = __builtin_shufflevector(lo, hi, 0, 1, 2, 3, 4, 5, 6, 7,
                                                    8, 9, 10, 11, 12, 13, 14, 15);
            }
#pragma unroll
            for (int fn = 0; fn < 4; ++fn) {
                // B 32x16: lane n=lr holds K {lh*8..+7} then {16+lh*8..+7}
                const _Float16* pb = &lds_b[(wn * 64 + fn * 16 + lr) * LDK + lh * 8];
                v8h lo = *(const v8h*)pb;
                v8h hi = *(const v8h*)(pb + 16);
                bfrag[fn] = __builtin_shufflevector(lo, hi, 0, 1, 2, 3, 4, 5, 6, 7,
                                                    8, 9, 10, 11, 12, 13, 14, 15);
            }
#pragma unroll
            for (int fm = 0; fm < 2; ++fm)
#pragma unroll
                for (int fn = 0; fn < 4; ++fn)
                    acc[fm][fn] = __builtin_amdgcn_wmma_f32_16x16x32_f16(
                        false, afrag[fm], false, bfrag[fn],
                        (short)0, acc[fm][fn], false, false);
        }
    }

    // ---- epilogue. C/D layout: element r -> (m = r + 8*lh, n = lr)
#pragma unroll
    for (int fn = 0; fn < 4; ++fn) {
        const int p  = ptile + wn * 64 + fn * 16 + lr;
        const int py = p >> 6, px = p & 63;
        const float nz = noise[b * NPIX + p];
#pragma unroll
        for (int fm = 0; fm < 2; ++fm) {
            const int obase = otile + wm * 32 + fm * 16 + lh * 8;
#pragma unroll
            for (int r = 0; r < 8; ++r) {
                const int o = obase + r;
                float v = acc[fm][fn][r];
                v = v * demod[b * CH + o] + bias[o] + nscale[o] * nz;
                v = (v >= 0.f) ? v : 0.2f * v;
                if (s_next) v *= s_next[b * CH + o];
                out[(size_t)(b * CH + o) * oplane + py * orow + px + ooff] = (OutT)v;
            }
        }
    }
}

// ---------------------------------------------------------------------------
// to_rgb: rgb_out = rgb_up + rgbB + sum_ci rgbW[c,ci]*s2[b,ci]*x[b,ci,p]
// ---------------------------------------------------------------------------
__global__ void torgb_kernel(const float* __restrict__ x, const float* __restrict__ s2,
                             const float* __restrict__ rgbW, const float* __restrict__ rgbB,
                             const float* __restrict__ rgbup, float* __restrict__ out) {
    int idx = blockIdx.x * blockDim.x + threadIdx.x;     // BATCH*NPIX
    if (idx >= BATCH * NPIX) return;
    int b = idx >> 12, p = idx & 4095;
    const float* xb = x + (size_t)b * CH * NPIX + p;
    const float* sb = s2 + (size_t)b * CH;
    float a0 = 0.f, a1 = 0.f, a2 = 0.f;
    for (int ci = 0; ci < CH; ++ci) {
        float xm = xb[(size_t)ci * NPIX] * sb[ci];
        a0 += rgbW[ci] * xm;
        a1 += rgbW[CH + ci] * xm;
        a2 += rgbW[2 * CH + ci] * xm;
    }
    out[((size_t)b * 3 + 0) * NPIX + p] = rgbup[((size_t)b * 3 + 0) * NPIX + p] + a0 + rgbB[0];
    out[((size_t)b * 3 + 1) * NPIX + p] = rgbup[((size_t)b * 3 + 1) * NPIX + p] + a1 + rgbB[1];
    out[((size_t)b * 3 + 2) * NPIX + p] = rgbup[((size_t)b * 3 + 2) * NPIX + p] + a2 + rgbB[2];
}

// ---------------------------------------------------------------------------
extern "C" void kernel_launch(void* const* d_in, const int* in_sizes, int n_in,
                              void* d_out, int out_size, void* d_ws, size_t ws_size,
                              hipStream_t stream) {
    const float* maps   = (const float*)d_in[0];
    const float* wlat   = (const float*)d_in[1];
    const float* rgb    = (const float*)d_in[2];
    const float* noise0 = (const float*)d_in[3];
    const float* noise1 = (const float*)d_in[4];
    const float* w0     = (const float*)d_in[5];
    const float* b0     = (const float*)d_in[6];
    const float* sW0    = (const float*)d_in[7];
    const float* sB0    = (const float*)d_in[8];
    const float* ns0    = (const float*)d_in[9];
    const float* w1     = (const float*)d_in[10];
    const float* b1     = (const float*)d_in[11];
    const float* sW1    = (const float*)d_in[12];
    const float* sB1    = (const float*)d_in[13];
    const float* ns1    = (const float*)d_in[14];
    const float* rgbW   = (const float*)d_in[15];
    const float* rgbB   = (const float*)d_in[16];
    const float* rgbSW  = (const float*)d_in[17];
    const float* rgbSB  = (const float*)d_in[18];

    float* out_rgb = (float*)d_out;                              // [8,3,64,64]
    float* out_x   = (float*)d_out + (size_t)BATCH * 3 * NPIX;   // [8,512,64,64]

    // ---- workspace layout (~84 MB)
    char* wsb = (char*)d_ws;
    const size_t ACTP_BYTES = (size_t)BATCH * CH * PPLANE * sizeof(_Float16); // 35,684,352
    const size_t WH_BYTES   = (size_t)9 * CH * CH * sizeof(_Float16);         // 4,718,592
    _Float16* X0p = (_Float16*)wsb;                      // padded, maps*s0
    _Float16* X1p = (_Float16*)(wsb + ACTP_BYTES);       // padded, layer0 out * s1
    _Float16* WH0 = (_Float16*)(wsb + 2 * ACTP_BYTES);
    _Float16* WH1 = (_Float16*)(wsb + 2 * ACTP_BYTES + WH_BYTES);
    float* RGBUP  = (float*)(wsb + 2 * ACTP_BYTES + 2 * WH_BYTES);
    float* S0     = RGBUP + (size_t)BATCH * 3 * NPIX;
    float* S1     = S0 + BATCH * CH;
    float* S2     = S1 + BATCH * CH;
    float* D0     = S2 + BATCH * CH;
    float* D1     = D0 + BATCH * CH;
    float* WSQ0   = D1 + BATCH * CH;                     // 262,144 floats
    float* WSQ1   = WSQ0 + (size_t)CH * CH;

    // ---- clear padded activation buffers (incl. zero borders) every call
    {
        int nwords = (int)(2 * ACTP_BYTES / 4);          // 17,842,176
        zero_kernel<<<(nwords + 255) / 256, 256, 0, stream>>>((uint32_t*)wsb, nwords);
    }

    // ---- one-time per call: f16 weight repack + styles + demod (all tiny)
    repack_w<<<(9 * CH * CH + 255) / 256, 256, 0, stream>>>(w0, WH0);
    repack_w<<<(9 * CH * CH + 255) / 256, 256, 0, stream>>>(w1, WH1);
    wsq_kernel<<<(CH * CH + 255) / 256, 256, 0, stream>>>(w0, WSQ0);
    wsq_kernel<<<(CH * CH + 255) / 256, 256, 0, stream>>>(w1, WSQ1);
    style_kernel<<<(BATCH * CH + 255) / 256, 256, 0, stream>>>(wlat, sW0, sB0, S0);
    style_kernel<<<(BATCH * CH + 255) / 256, 256, 0, stream>>>(wlat, sW1, sB1, S1);
    style_kernel<<<(BATCH * CH + 255) / 256, 256, 0, stream>>>(wlat, rgbSW, rgbSB, S2);
    demod_kernel<<<(BATCH * CH + 255) / 256, 256, 0, stream>>>(WSQ0, S0, D0);
    demod_kernel<<<(BATCH * CH + 255) / 256, 256, 0, stream>>>(WSQ1, S1, D1);

    // ---- upsample: maps (style-modulated, f16 padded) and rgb (f32 dense)
    {
        int total = BATCH * CH * NPIX;
        upsample2x_mod<_Float16><<<(total + 255) / 256, 256, 0, stream>>>(
            maps, S0, X0p, CH, PPLANE, PROW, PROW + 1);
    }
    {
        int total = BATCH * 3 * NPIX;
        upsample2x_mod<float><<<(total + 255) / 256, 256, 0, stream>>>(
            rgb, nullptr, RGBUP, 3, NPIX, 64, 0);
    }

    // ---- WMMA conv layers
    dim3 cgrid(NPIX / 128, CH / 128, BATCH);             // (32, 4, 8)
    // layer 0: X0p -> X1p (padded interior), epilogue pre-applies style S1
    conv3x3_wmma<_Float16><<<cgrid, 256, 0, stream>>>(
        X0p, WH0, D0, b0, ns0, noise0, S1, X1p, PPLANE, PROW, PROW + 1);
    // layer 1: X1p -> out_x (dense f32, returned output)
    conv3x3_wmma<float><<<cgrid, 256, 0, stream>>>(
        X1p, WH1, D1, b1, ns1, noise1, nullptr, out_x, NPIX, 64, 0);

    // ---- to_rgb
    torgb_kernel<<<(BATCH * NPIX + 255) / 256, 256, 0, stream>>>(
        out_x, S2, rgbW, rgbB, RGBUP, out_rgb);
}
